// InstrumentQuantizer_83107617178091
// MI455X (gfx1250) — matrix operand order repaired and checked by reading, not compiled, not run on hardware
//
#include <hip/hip_runtime.h>
#include <hip/hip_bf16.h>
#include <stdint.h>

#define B_      16
#define S_      1024
#define NTOK    16384      /* B_*S_ */
#define DIM     512
#define OPD     128
#define INNER   1024
#define CLASSES 512
#define CBD     32
#define SEQL    5
#define HEADS   4

typedef __bf16 bf16;
typedef __attribute__((ext_vector_type(16))) __bf16 bf16x16;
typedef __attribute__((ext_vector_type(8)))  float  f32x8;
typedef __attribute__((ext_vector_type(4)))  unsigned uint32x4;
typedef __attribute__((ext_vector_type(8)))  int      int32x8;
typedef __attribute__((ext_vector_type(4)))  int      int32x4;

// native f32 -> bf16 convert (lowers to v_cvt_pk_bf16_f32 on gfx1250)
__device__ __forceinline__ bf16 f2bf(float x) { return (bf16)x; }

__device__ __forceinline__ float geluf(float x) {
  // tanh approximation (jax.nn.gelu default, approximate=True)
  float x3 = x * x * x;
  return 0.5f * x * (1.0f + tanhf(0.7978845608028654f * (x + 0.044715f * x3)));
}

// LDS byte offset of a __shared__ object (flat LDS aperture: low 32 bits)
__device__ __forceinline__ unsigned lds_off(const void* p) {
  return (unsigned)(size_t)p;
}

// async copy of 16 bytes global -> LDS (ASYNCcnt-tracked, no VGPR round trip)
__device__ __forceinline__ void async_ld_b128(unsigned ldsAddr, const float* gaddr) {
  asm volatile("global_load_async_to_lds_b128 %0, %1, off"
               :: "v"(ldsAddr), "v"(gaddr) : "memory");
}
__device__ __forceinline__ void wait_async0() {
  asm volatile("s_wait_asynccnt 0x0" ::: "memory");
}

// Tensor Data Mover: DMA a tileRows x tileCols f32 tile (row stride = ldRow
// elements) from global memory into LDS (row-major, compacted).
// D# layout per CDNA5 ISA 8.3/8.4: count=1, type=2 ("image"), data_size=4B.
// This toolchain exposes the 6-arg builtin (extra int32x8 group before cpol).
__device__ __forceinline__ void tdm_load_tile_f32(
    unsigned ldsAddr, const float* gtile, int tileCols, int tileRows,
    int ldRow /*elements*/, int tensorRows)
{
  unsigned long long ga = (unsigned long long)(size_t)gtile;
  uint32x4 g0;
  g0[0] = 1u;                                   // count=1, user descriptor
  g0[1] = ldsAddr;                              // lds_addr (bytes)
  g0[2] = (unsigned)ga;                         // global_addr[31:0]
  g0[3] = (unsigned)(ga >> 32) | (2u << 30);    // global_addr[56:32] | type=2
  int32x8 g1;
  g1[0] = 0x00020000;                           // data_size=2 (4 bytes)
  g1[1] = (int)(((unsigned)ldRow & 0xffffu) << 16);            // tensor_dim0 lo
  g1[2] = (int)(((unsigned)ldRow >> 16) |
                (((unsigned)tensorRows & 0xffffu) << 16));     // dim0 hi|dim1 lo
  g1[3] = (int)(((unsigned)tensorRows >> 16) |
                ((unsigned)tileCols << 16));                   // dim1 hi|tile_dim0
  g1[4] = (int)((unsigned)tileRows);            // tile_dim1 (tile_dim2=0)
  g1[5] = ldRow;                                // tensor_dim0_stride[31:0]
  g1[6] = 0;                                    // stride0 hi | stride1 lo
  g1[7] = 0;                                    // stride1 hi
  int32x4 gz4 = {0, 0, 0, 0};                   // groups 2/3: dims 2..4 unused
  int32x8 gz8 = {0, 0, 0, 0, 0, 0, 0, 0};
  __builtin_amdgcn_tensor_load_to_lds(g0, g1, gz4, gz4, gz8, 0);
}

// ---------------------------------------------------------------------------
// Generic WMMA GEMM: C[M,N] = A[M,K] @ B[K,N] (+bias[N]) (+res[M,N])
// Row-major, lda=K, ldb=N, ldc=N. M%128==0, N%128==0, K%32==0.
// Block tile 128x128, BK=32, 8 waves (4M x 2N), wave tile 32x64.
// A tile DMA'd by the Tensor Data Mover (wave 0 issues, TENSORcnt-tracked);
// B tile staged manually (transpose + bf16 convert).
// ---------------------------------------------------------------------------
__global__ __launch_bounds__(256) void gemm_kernel(
    const float* __restrict__ A, const float* __restrict__ B,
    const float* __restrict__ bias, const float* __restrict__ res,
    float* __restrict__ C, int M, int N, int K)
{
  __shared__ float AsF[128][32];     // TDM-staged A tile (f32, row-major)
  __shared__ bf16  Bs[128][34];      // B tile transposed: Bs[n][k] (bf16)
  const int tid  = threadIdx.x;
  const int bm   = blockIdx.y * 128;
  const int bn   = blockIdx.x * 128;
  const int wave = tid >> 5, lane = tid & 31;
  const int wm   = (wave & 3) * 32;
  const int wn   = (wave >> 2) * 64;
  const int half = lane >> 4, r = lane & 15;

  const unsigned ldsA = lds_off(&AsF[0][0]);
  const float*   aTile = A + (size_t)bm * K;          // advances by +32 per step

  // per-thread B staging pointers (advance by 32*N per step)
  const float* bP[4];
  #pragma unroll
  for (int it = 0; it < 4; ++it)
    bP[it] = B + (size_t)((tid >> 5) + it * 8) * N + bn + (tid & 31) * 4;

  f32x8 acc[2][4] = {};

  for (int k0 = 0; k0 < K; k0 += 32) {
    // one TDM descriptor moves the whole 128x32 f32 A tile into LDS
    if (wave == 0) {
      tdm_load_tile_f32(ldsA, aTile, /*tileCols=*/32, /*tileRows=*/128,
                        /*ldRow=*/K, /*tensorRows=*/M);
    }
    // stage B tile (32x128 f32 -> bf16, transposed into [n][k])
    #pragma unroll
    for (int it = 0; it < 4; ++it) {
      int kk  = (tid >> 5) + it * 8;
      int col = (tid & 31) * 4;
      const float4 v = *(const float4*)bP[it];
      Bs[col    ][kk] = f2bf(v.x);
      Bs[col + 1][kk] = f2bf(v.y);
      Bs[col + 2][kk] = f2bf(v.z);
      Bs[col + 3][kk] = f2bf(v.w);
      bP[it] += (size_t)32 * N;
    }
    if (k0 + 32 < K)   // hint next B K-tile toward the caches
      __builtin_prefetch(bP[0], 0, 1);

    if (wave == 0) __builtin_amdgcn_s_wait_tensorcnt(0);
    __syncthreads();   // TDM tile + all waves' B stores visible

    bf16x16 af[2], bfr[4];
    #pragma unroll
    for (int i = 0; i < 2; ++i) {          // ISA A layout: K pairs interleaved
      const float* p = &AsF[wm + i * 16 + r][0];
      #pragma unroll
      for (int e = 0; e < 16; ++e)
        af[i][e] = f2bf(p[((e < 8) ? e : e + 8) + 8 * half]);
    }
    #pragma unroll
    for (int j = 0; j < 4; ++j) {          // ISA B layout: K pairs interleaved (same as A)
      const bf16* p = &Bs[wn + j * 16 + r][0];
      #pragma unroll
      for (int e = 0; e < 16; ++e)
        bfr[j][e] = p[((e < 8) ? e : e + 8) + 8 * half];
    }
    #pragma unroll
    for (int i = 0; i < 2; ++i)
      #pragma unroll
      for (int j = 0; j < 4; ++j)
        acc[i][j] = __builtin_amdgcn_wmma_f32_16x16x32_bf16(
            false, af[i], false, bfr[j], (short)0, acc[i][j], false, false);
    __syncthreads();
    aTile += 32;
  }

  #pragma unroll
  for (int i = 0; i < 2; ++i)
    #pragma unroll
    for (int j = 0; j < 4; ++j) {
      int   col = bn + wn + j * 16 + r;
      float bb  = bias ? bias[col] : 0.0f;
      #pragma unroll
      for (int e = 0; e < 8; ++e) {
        int    row = bm + wm + i * 16 + 8 * half + e;   // C layout: M=e+8*half
        size_t off = (size_t)row * N + col;
        float  v   = acc[i][j][e] + bb;
        if (res) v += res[off];
        C[off] = v;
      }
    }
}

// ---------------------------------------------------------------------------
// Fused GLU GEMM: Out[M,inner] = (A@W[:,n]+b1[n]) * gelu(A@W[:,inner+n]+b1[inner+n])
// W is [K, 2*inner] row-major. Block tile 128x64, waves 4M x 2N (32x32 each).
// A tile staged via GLOBAL_LOAD_ASYNC_TO_LDS_B128 (ASYNCcnt-tracked).
// ---------------------------------------------------------------------------
__global__ __launch_bounds__(256) void gemm_glu_kernel(
    const float* __restrict__ A, const float* __restrict__ W,
    const float* __restrict__ b1, float* __restrict__ Out,
    int M, int K, int inner)
{
  __shared__ float AsF[128][32];
  __shared__ bf16  BsA[64][34];
  __shared__ bf16  BsG[64][34];
  const int ldw  = 2 * inner;
  const int tid  = threadIdx.x;
  const int bm   = blockIdx.y * 128;
  const int bn   = blockIdx.x * 64;
  const int wave = tid >> 5, lane = tid & 31;
  const int wm   = (wave & 3) * 32;
  const int wn   = (wave >> 2) * 32;
  const int half = lane >> 4, r = lane & 15;

  const int arow = tid >> 3;         // 0..31
  const int acol = (tid & 7) * 4;    // 0..28 (float index)

  // precompute per-thread addresses; advance per K-step (no muls in loop)
  unsigned     ldsAddr[4];
  const float* aP[4];
  #pragma unroll
  for (int it = 0; it < 4; ++it) {
    int row = arow + it * 32;
    ldsAddr[it] = lds_off(&AsF[0][0]) + (unsigned)(row * 32 + acol) * 4u;
    aP[it]      = A + (size_t)(bm + row) * K + acol;
  }
  const float* wPA[2];
  const float* wPG[2];
  #pragma unroll
  for (int it = 0; it < 2; ++it) {
    int kk  = (tid >> 4) + it * 16;
    int col = (tid & 15) * 4;
    wPA[it] = W + (size_t)kk * ldw + bn + col;
    wPG[it] = W + (size_t)kk * ldw + inner + bn + col;
  }

  f32x8 accA[2][2] = {};
  f32x8 accG[2][2] = {};

  for (int k0 = 0; k0 < K; k0 += 32) {
    #pragma unroll
    for (int it = 0; it < 4; ++it) {
      async_ld_b128(ldsAddr[it], aP[it]);
      aP[it] += 32;
    }
    #pragma unroll
    for (int it = 0; it < 2; ++it) {
      int kk  = (tid >> 4) + it * 16;
      int col = (tid & 15) * 4;
      const float4 va = *(const float4*)wPA[it];
      const float4 vg = *(const float4*)wPG[it];
      BsA[col][kk] = f2bf(va.x); BsA[col + 1][kk] = f2bf(va.y);
      BsA[col + 2][kk] = f2bf(va.z); BsA[col + 3][kk] = f2bf(va.w);
      BsG[col][kk] = f2bf(vg.x); BsG[col + 1][kk] = f2bf(vg.y);
      BsG[col + 2][kk] = f2bf(vg.z); BsG[col + 3][kk] = f2bf(vg.w);
      wPA[it] += (size_t)32 * ldw;
      wPG[it] += (size_t)32 * ldw;
    }
    wait_async0();
    __syncthreads();

    bf16x16 af[2], bA[2], bG[2];
    #pragma unroll
    for (int i = 0; i < 2; ++i) {
      const float* p = &AsF[wm + i * 16 + r][0];
      #pragma unroll
      for (int e = 0; e < 16; ++e)
        af[i][e] = f2bf(p[((e < 8) ? e : e + 8) + 8 * half]);
    }
    #pragma unroll
    for (int j = 0; j < 2; ++j) {
      const bf16* pa = &BsA[wn + j * 16 + r][0];
      const bf16* pg = &BsG[wn + j * 16 + r][0];
      #pragma unroll
      for (int e = 0; e < 16; ++e) { bA[j][e] = pa[((e < 8) ? e : e + 8) + 8 * half]; bG[j][e] = pg[((e < 8) ? e : e + 8) + 8 * half]; }
    }
    #pragma unroll
    for (int i = 0; i < 2; ++i)
      #pragma unroll
      for (int j = 0; j < 2; ++j) {
        accA[i][j] = __builtin_amdgcn_wmma_f32_16x16x32_bf16(
            false, af[i], false, bA[j], (short)0, accA[i][j], false, false);
        accG[i][j] = __builtin_amdgcn_wmma_f32_16x16x32_bf16(
            false, af[i], false, bG[j], (short)0, accG[i][j], false, false);
      }
    __syncthreads();
  }

  #pragma unroll
  for (int i = 0; i < 2; ++i)
    #pragma unroll
    for (int j = 0; j < 2; ++j) {
      int   col = bn + wn + j * 16 + r;
      float ba = b1[col], bg = b1[inner + col];
      #pragma unroll
      for (int e = 0; e < 8; ++e) {
        int row = bm + wm + i * 16 + 8 * half + e;
        float a = accA[i][j][e] + ba;
        float g = accG[i][j][e] + bg;
        Out[(size_t)row * inner + col] = a * geluf(g);
      }
    }
}

// ---------------------------------------------------------------------------
// Elementwise / small kernels
// ---------------------------------------------------------------------------
__global__ void transpose_x_kernel(const float* __restrict__ x, float* __restrict__ px) {
  int idx = blockIdx.x * 256 + threadIdx.x;          // over B_*OPD*S_
  int s = idx % S_;
  int c = (idx / S_) % OPD;
  int b = idx / (S_ * OPD);
  px[((size_t)(b * S_ + s)) * OPD + c] = x[idx];
}

__global__ void bn_kernel(const float* __restrict__ in, float* __restrict__ out,
                          const float* __restrict__ g, const float* __restrict__ b,
                          int rows, int C) {
  int c = blockIdx.x, tid = threadIdx.x;
  float s = 0.f, s2 = 0.f;
  for (int rr = tid; rr < rows; rr += 256) {
    float v = in[(size_t)rr * C + c]; s += v; s2 += v * v;
  }
  __shared__ float sh[256], sh2[256];
  sh[tid] = s; sh2[tid] = s2; __syncthreads();
  for (int o = 128; o > 0; o >>= 1) {
    if (tid < o) { sh[tid] += sh[tid + o]; sh2[tid] += sh2[tid + o]; }
    __syncthreads();
  }
  float mean = sh[0] / rows;
  float var  = sh2[0] / rows - mean * mean;
  float sc = rsqrtf(var + 1e-5f) * g[c];
  float sb = b[c] - mean * sc;
  for (int rr = tid; rr < rows; rr += 256) {
    size_t o = (size_t)rr * C + c;
    out[o] = in[o] * sc + sb;
  }
}

__global__ void rms_kernel(const float* __restrict__ in, float* __restrict__ out,
                           const float* __restrict__ g, int rows,
                           int inStride, int outStride) {
  int w = blockIdx.x * 8 + (threadIdx.x >> 5);
  int lane = threadIdx.x & 31;
  if (w >= rows) return;
  const float* ip = in + (size_t)w * inStride;
  float v[16]; float s = 0.f;
  #pragma unroll
  for (int i = 0; i < 16; ++i) { v[i] = ip[lane + i * 32]; s += v[i] * v[i]; }
  #pragma unroll
  for (int o = 16; o > 0; o >>= 1) s += __shfl_xor(s, o, 32);
  float sc = rsqrtf(s + 1e-8f) * 22.62741699796952f;   // sqrt(512)
  float* op = out + (size_t)w * outStride;
  #pragma unroll
  for (int i = 0; i < 16; ++i) op[lane + i * 32] = v[i] * sc * g[lane + i * 32];
}

__global__ void attn_kernel(const float* __restrict__ Q, const float* __restrict__ K,
                            const float* __restrict__ V, float* __restrict__ O, int t) {
  int w = blockIdx.x * 8 + (threadIdx.x >> 5);       // one wave per (n, head)
  int lane = threadIdx.x & 31;
  if (w >= NTOK * HEADS) return;
  int n = w >> 2, h = w & 3;
  size_t base = ((size_t)n * t) * DIM + h * 128 + lane * 4;
  float q[SEQL][4], k[SEQL][4], v[SEQL][4];
  for (int i = 0; i < t; ++i) {
    float4 a = *(const float4*)(Q + base + (size_t)i * DIM);
    float4 b = *(const float4*)(K + base + (size_t)i * DIM);
    float4 c = *(const float4*)(V + base + (size_t)i * DIM);
    q[i][0]=a.x; q[i][1]=a.y; q[i][2]=a.z; q[i][3]=a.w;
    k[i][0]=b.x; k[i][1]=b.y; k[i][2]=b.z; k[i][3]=b.w;
    v[i][0]=c.x; v[i][1]=c.y; v[i][2]=c.z; v[i][3]=c.w;
  }
  const float scale = 0.08838834764831845f;          // 1/sqrt(128)
  float sc[SEQL][SEQL];
  for (int i = 0; i < t; ++i)
    for (int j = 0; j <= i; ++j) {
      float s = q[i][0]*k[j][0] + q[i][1]*k[j][1] + q[i][2]*k[j][2] + q[i][3]*k[j][3];
      #pragma unroll
      for (int o = 16; o > 0; o >>= 1) s += __shfl_xor(s, o, 32);
      sc[i][j] = s * scale;
    }
  float ov[SEQL][4] = {};
  for (int i = 0; i < t; ++i) {
    float m = -3.0e38f;
    for (int j = 0; j <= i; ++j) m = fmaxf(m, sc[i][j]);
    float den = 0.f;
    for (int j = 0; j <= i; ++j) { float e = __expf(sc[i][j] - m); den += e; sc[i][j] = e; }
    float inv = 1.0f / den;
    for (int j = 0; j <= i; ++j) {
      float a = sc[i][j] * inv;
      ov[i][0] += a * v[j][0]; ov[i][1] += a * v[j][1];
      ov[i][2] += a * v[j][2]; ov[i][3] += a * v[j][3];
    }
  }
  for (int i = 0; i < t; ++i) {
    float4 rr; rr.x = ov[i][0]; rr.y = ov[i][1]; rr.z = ov[i][2]; rr.w = ov[i][3];
    *(float4*)(O + base + (size_t)i * DIM) = rr;
  }
}

__global__ void prep_en_kernel(const float* __restrict__ embed, float* __restrict__ en) {
  int row = blockIdx.x * 8 + (threadIdx.x >> 5);
  int lane = threadIdx.x & 31;
  if (row >= CLASSES) return;
  float v = (row == 0) ? 0.0f : embed[(size_t)row * CBD + lane];
  float s = v * v;
  #pragma unroll
  for (int o = 16; o > 0; o >>= 1) s += __shfl_xor(s, o, 32);
  en[(size_t)row * CBD + lane] = v * rsqrtf(s + 1e-10f);
}

__global__ void zproj_kernel(const float* __restrict__ dlast, const float* __restrict__ pin_w,
                             const float* __restrict__ pin_b, float* __restrict__ z) {
  int i = blockIdx.x * 256 + threadIdx.x;            // N*32
  int n = i >> 5, j = i & 31;
  float s = pin_b[j];
  const float* dr = dlast + (size_t)n * DIM;
  for (int kk = 0; kk < DIM; ++kk) s += dr[kk] * pin_w[kk * CBD + j];
  z[i] = s;
}

__global__ void vq_argmax_kernel(const float* __restrict__ z, const float* __restrict__ en,
                                 unsigned* __restrict__ mask, int* __restrict__ idxOut) {
  __shared__ float zs[8][CBD];
  int wv = threadIdx.x >> 5, lane = threadIdx.x & 31;
  int n = blockIdx.x * 8 + wv;
  float v = z[(size_t)n * CBD + lane];
  float s = v * v;
  #pragma unroll
  for (int o = 16; o > 0; o >>= 1) s += __shfl_xor(s, o, 32);
  zs[wv][lane] = v * rsqrtf(s + 1e-10f);
  __syncthreads();
  float best = -3.0e38f; int bi = 0x7fffffff;
  for (int it = 0; it < 16; ++it) {
    int c = it * 32 + lane;
    const float* e = en + (size_t)c * CBD;
    float sm = 0.f;
    #pragma unroll
    for (int d = 0; d < CBD; ++d) sm += zs[wv][d] * e[d];
    if ((mask[(size_t)n * 16 + it] >> lane) & 1u) sm = -1e9f;
    if (sm > best || (sm == best && c < bi)) { best = sm; bi = c; }
  }
  #pragma unroll
  for (int o = 16; o > 0; o >>= 1) {
    float ob = __shfl_xor(best, o, 32);
    int   oi = __shfl_xor(bi, o, 32);
    if (ob > best || (ob == best && oi < bi)) { best = ob; bi = oi; }
  }
  if (lane == 0) {
    idxOut[n] = bi;
    if (bi != 0) mask[(size_t)n * 16 + (bi >> 5)] |= (1u << (bi & 31));  // code 0 -> -1 -> no bit
  }
}

__global__ void gather_en_kernel(const float* __restrict__ en, const int* __restrict__ idx,
                                 float* __restrict__ zsel) {
  int i = blockIdx.x * 256 + threadIdx.x;            // N*32
  int n = i >> 5, d = i & 31;
  zsel[i] = en[(size_t)idx[n] * CBD + d];
}

__global__ void stack_write_kernel(const float* __restrict__ src, float* __restrict__ stack, int slot) {
  int idx = blockIdx.x * 256 + threadIdx.x;          // N*512
  int n = idx >> 9, c = idx & 511;
  stack[((size_t)n * SEQL + slot) * DIM + c] = src[idx];
}

__global__ void build_seq_kernel(const float* __restrict__ stack, float* __restrict__ x, int t) {
  int idx = blockIdx.x * 256 + threadIdx.x;          // N*t*512
  if (idx >= NTOK * t * DIM) return;
  int c = idx & 511;
  int tmp = idx >> 9;
  int tok = tmp % t;
  int n = tmp / t;
  x[idx] = stack[((size_t)n * SEQL + tok) * DIM + c];
}

__global__ void loss_kernel(const float* __restrict__ cum, const float* __restrict__ px,
                            float* __restrict__ acc) {
  __shared__ float sh[256];
  int tid = threadIdx.x;
  const size_t total = (size_t)NTOK * OPD;
  float s = 0.f;
  for (size_t p = (size_t)blockIdx.x * 256 + tid; p < total; p += (size_t)gridDim.x * 256) {
    float d = cum[p] - px[p]; s += d * d;
  }
  sh[tid] = s; __syncthreads();
  for (int o = 128; o > 0; o >>= 1) { if (tid < o) sh[tid] += sh[tid + o]; __syncthreads(); }
  if (tid == 0) atomicAdd(acc, sh[0]);
}

__global__ void finalize_kernel(const float* __restrict__ lacc, float* __restrict__ out) {
  int j = threadIdx.x;
  if (j < SEQL)
    out[j] = lacc[j] * (1.0f / ((float)NTOK * (float)OPD)) * ((float)(j + 1) / (float)SEQL);
}

// ---------------------------------------------------------------------------
// Host orchestration
// ---------------------------------------------------------------------------
extern "C" void kernel_launch(void* const* d_in, const int* in_sizes, int n_in,
                              void* d_out, int out_size, void* d_ws, size_t ws_size,
                              hipStream_t stream) {
  (void)in_sizes; (void)n_in; (void)out_size; (void)ws_size;
  // input order = setup_inputs() insertion order
  const float* x       = (const float*)d_in[0];
  const float* proj_w  = (const float*)d_in[1];
  const float* proj_b  = (const float*)d_in[2];
  // enc layer l: d_in[3 + 6l + {bn_g, bn_b, w1, b1, w2, b2}]
  const float* fbn_g   = (const float*)d_in[27];
  const float* fbn_b   = (const float*)d_in[28];
  // dec layer l: d_in[29 + 10l + {attn_g, wq, wk, wv, wo, ff_g, w1, b1, w2, b2}]
  const float* final_g = (const float*)d_in[49];
  const float* embed   = (const float*)d_in[50];
  const float* pin_w   = (const float*)d_in[51];
  const float* pin_b   = (const float*)d_in[52];
  const float* pout_w  = (const float*)d_in[53];
  const float* pout_b  = (const float*)d_in[54];
  const float* out_w   = (const float*)d_in[55];
  const float* out_b   = (const float*)d_in[56];

  char* ws = (char*)d_ws;
  size_t off = 0;
  auto alloc = [&](size_t bytes) -> char* {
    char* p = ws + off;
    off = (off + bytes + 255) & ~(size_t)255;
    return p;
  };
  float*    px    = (float*)alloc((size_t)NTOK * OPD * 4);
  float*    stack = (float*)alloc((size_t)NTOK * SEQL * DIM * 4);
  float*    cum   = (float*)alloc((size_t)NTOK * OPD * 4);
  unsigned* mask  = (unsigned*)alloc((size_t)NTOK * 16 * 4);
  float*    en    = (float*)alloc((size_t)CLASSES * CBD * 4);
  int*      vidx  = (int*)alloc((size_t)NTOK * 4);
  float*    z     = (float*)alloc((size_t)NTOK * CBD * 4);
  float*    zsel  = (float*)alloc((size_t)NTOK * CBD * 4);
  float*    dlast = (float*)alloc((size_t)NTOK * DIM * 4);
  float*    qbuf  = (float*)alloc((size_t)NTOK * DIM * 4);
  float*    lacc  = (float*)alloc(64);
  float*    bufA  = (float*)alloc((size_t)NTOK * SEQL * DIM * 4);
  float*    bufB  = (float*)alloc((size_t)NTOK * SEQL * DIM * 4);
  float*    bufQ  = (float*)alloc((size_t)NTOK * SEQL * DIM * 4);
  float*    bufK  = (float*)alloc((size_t)NTOK * SEQL * DIM * 4);
  float*    bufV  = (float*)alloc((size_t)NTOK * SEQL * DIM * 4);
  float*    bufC  = (float*)alloc((size_t)NTOK * SEQL * INNER * 4);

  (void)hipMemsetAsync(cum, 0, (size_t)NTOK * OPD * 4, stream);
  (void)hipMemsetAsync(mask, 0, (size_t)NTOK * 16 * 4, stream);
  (void)hipMemsetAsync(lacc, 0, 64, stream);

  auto gemm = [&](const float* A, const float* Bm, const float* bias, const float* res,
                  float* C, int M, int N, int K) {
    dim3 g((unsigned)(N / 128), (unsigned)(M / 128));
    gemm_kernel<<<g, 256, 0, stream>>>(A, Bm, bias, res, C, M, N, K);
  };

  // ------------------ encoder ------------------
  transpose_x_kernel<<<(B_ * OPD * S_) / 256, 256, 0, stream>>>(x, px);
  gemm(px, proj_w, proj_b, nullptr, bufA, NTOK, DIM, OPD);
  for (int l = 0; l < 4; ++l) {
    const float* bn_g = (const float*)d_in[3 + l * 6 + 0];
    const float* bn_b = (const float*)d_in[3 + l * 6 + 1];
    const float* w1   = (const float*)d_in[3 + l * 6 + 2];
    const float* b1   = (const float*)d_in[3 + l * 6 + 3];
    const float* w2   = (const float*)d_in[3 + l * 6 + 4];
    const float* b2   = (const float*)d_in[3 + l * 6 + 5];
    bn_kernel<<<DIM, 256, 0, stream>>>(bufA, bufB, bn_g, bn_b, NTOK, DIM);
    dim3 gg(INNER / 64, NTOK / 128);
    gemm_glu_kernel<<<gg, 256, 0, stream>>>(bufB, w1, b1, bufC, NTOK, DIM, INNER);
    gemm(bufC, w2, b2, bufA, bufA, NTOK, DIM, INNER);
  }
  bn_kernel<<<DIM, 256, 0, stream>>>(bufA, bufA, fbn_g, fbn_b, NTOK, DIM);
  stack_write_kernel<<<(NTOK * DIM) / 256, 256, 0, stream>>>(bufA, stack, 0);
  prep_en_kernel<<<CLASSES / 8, 256, 0, stream>>>(embed, en);

  // ------------------ autoregressive loop ------------------
  for (int step = 1; step <= SEQL; ++step) {
    int t = step;
    int rows = NTOK * t;
    build_seq_kernel<<<(rows * DIM) / 256, 256, 0, stream>>>(stack, bufA, t);
    for (int l = 0; l < 2; ++l) {
      const float* attn_g = (const float*)d_in[29 + l * 10 + 0];
      const float* wq     = (const float*)d_in[29 + l * 10 + 1];
      const float* wk     = (const float*)d_in[29 + l * 10 + 2];
      const float* wv     = (const float*)d_in[29 + l * 10 + 3];
      const float* wo     = (const float*)d_in[29 + l * 10 + 4];
      const float* ff_g   = (const float*)d_in[29 + l * 10 + 5];
      const float* w1     = (const float*)d_in[29 + l * 10 + 6];
      const float* b1     = (const float*)d_in[29 + l * 10 + 7];
      const float* w2     = (const float*)d_in[29 + l * 10 + 8];
      const float* b2     = (const float*)d_in[29 + l * 10 + 9];
      rms_kernel<<<rows / 8, 256, 0, stream>>>(bufA, bufB, attn_g, rows, DIM, DIM);
      gemm(bufB, wq, nullptr, nullptr, bufQ, rows, DIM, DIM);
      gemm(bufB, wk, nullptr, nullptr, bufK, rows, DIM, DIM);
      gemm(bufB, wv, nullptr, nullptr, bufV, rows, DIM, DIM);
      attn_kernel<<<(NTOK * HEADS) / 8, 256, 0, stream>>>(bufQ, bufK, bufV, bufB, t);
      gemm(bufB, wo, nullptr, bufA, bufA, rows, DIM, DIM);
      rms_kernel<<<rows / 8, 256, 0, stream>>>(bufA, bufB, ff_g, rows, DIM, DIM);
      dim3 gg(INNER / 64, rows / 128);
      gemm_glu_kernel<<<gg, 256, 0, stream>>>(bufB, w1, b1, bufC, rows, DIM, INNER);
      gemm(bufC, w2, b2, bufA, bufA, rows, DIM, INNER);
    }
    // final rmsnorm, last token only -> compact [N, 512]
    rms_kernel<<<NTOK / 8, 256, 0, stream>>>(bufA + (size_t)(t - 1) * DIM, dlast, final_g,
                                             NTOK, t * DIM, DIM);
    zproj_kernel<<<(NTOK * CBD) / 256, 256, 0, stream>>>(dlast, pin_w, pin_b, z);
    vq_argmax_kernel<<<NTOK / 8, 256, 0, stream>>>(z, en, mask, vidx);
    gather_en_kernel<<<(NTOK * CBD) / 256, 256, 0, stream>>>(en, vidx, zsel);
    gemm(zsel, pout_w, pout_b, nullptr, qbuf, NTOK, DIM, CBD);   // q (vq forward value)
    if (step < SEQL)
      stack_write_kernel<<<(NTOK * DIM) / 256, 256, 0, stream>>>(qbuf, stack, step);
    gemm(qbuf, out_w, out_b, cum, cum, NTOK, OPD, DIM);          // cum += q@out_w + out_b
    loss_kernel<<<1024, 256, 0, stream>>>(cum, px, lacc + (step - 1));
  }
  finalize_kernel<<<1, 32, 0, stream>>>(lacc, (float*)d_out);
}
